// GraphModule_37538014167218
// MI455X (gfx1250) — compile-verified, matrix operand order checked by reading, not run on hardware
//
#include <hip/hip_runtime.h>
#include <hip/hip_fp16.h>
#include <math.h>

typedef __attribute__((ext_vector_type(16))) _Float16 v16h;
typedef __attribute__((ext_vector_type(4)))  _Float16 v4h;
typedef __attribute__((ext_vector_type(8)))  float    v8f;

#define B_    32
#define C_    128
#define NH_   4
#define HD_   32
#define WW_   64
#define BW_   2048
#define HW_   4096
#define NTOK  (B_*HW_)     /* 131072 rows */
#define FF_   512
#define LN_EPS 1e-5f

// ---------------------------------------------------------------- WMMA helpers
__device__ __forceinline__ v8f wmma_f16(v16h a, v16h b, v8f c) {
  return __builtin_amdgcn_wmma_f32_16x16x32_f16(false, a, false, b, (short)0, c,
                                                false, false);
}

// Load 16 contiguous fp32 values and convert to one K-half of a 16x32 fragment.
__device__ __forceinline__ v16h frag_f32(const float* p) {
  const float4* q = (const float4*)p;
  float4 x0 = q[0], x1 = q[1], x2 = q[2], x3 = q[3];
  v16h v;
  v[0]=(_Float16)x0.x;  v[1]=(_Float16)x0.y;  v[2]=(_Float16)x0.z;  v[3]=(_Float16)x0.w;
  v[4]=(_Float16)x1.x;  v[5]=(_Float16)x1.y;  v[6]=(_Float16)x1.z;  v[7]=(_Float16)x1.w;
  v[8]=(_Float16)x2.x;  v[9]=(_Float16)x2.y;  v[10]=(_Float16)x2.z; v[11]=(_Float16)x2.w;
  v[12]=(_Float16)x3.x; v[13]=(_Float16)x3.y; v[14]=(_Float16)x3.z; v[15]=(_Float16)x3.w;
  return v;
}

// Direct 32B fp16 fragment load (global or LDS), no conversion VALU.
__device__ __forceinline__ v16h frag_h(const _Float16* p) {
  return *(const v16h*)p;
}

// ------------------------------------------- one-time fp32 -> fp16 weight cast
__global__ __launch_bounds__(256)
void k_cvt_f16(const float4* __restrict__ s, v4h* __restrict__ d) {
  int i = blockIdx.x * 256 + threadIdx.x;
  float4 v = s[i];
  v4h o;
  o[0] = (_Float16)v.x; o[1] = (_Float16)v.y;
  o[2] = (_Float16)v.z; o[3] = (_Float16)v.w;
  d[i] = o;
}

// ---------------------------------------------------- CPB bias table (tiny MLP)
__global__ __launch_bounds__(256)
void k_cpb_table(const float* __restrict__ rct, const float* __restrict__ w0,
                 const float* __restrict__ b0, const float* __restrict__ w2,
                 float* __restrict__ table) {
  __shared__ float red[4 * 256];
  const int tid = threadIdx.x;
  const int row = blockIdx.x;                 // 0..224
  const float t0 = rct[row * 2 + 0], t1 = rct[row * 2 + 1];
  float pp[4] = {0.f, 0.f, 0.f, 0.f};
  for (int u = tid; u < 512; u += 256) {
    float h = fmaxf(t0 * w0[u * 2 + 0] + t1 * w0[u * 2 + 1] + b0[u], 0.f);
#pragma unroll
    for (int hh = 0; hh < 4; ++hh) pp[hh] += h * w2[hh * 512 + u];
  }
#pragma unroll
  for (int hh = 0; hh < 4; ++hh) red[hh * 256 + tid] = pp[hh];
  __syncthreads();
  for (int s = 128; s >= 1; s >>= 1) {
    if (tid < s) {
#pragma unroll
      for (int hh = 0; hh < 4; ++hh) red[hh * 256 + tid] += red[hh * 256 + tid + s];
    }
    __syncthreads();
  }
  if (tid < 4) table[row * 4 + tid] = red[tid * 256];
}

__global__ __launch_bounds__(256)
void k_bias_expand(const float* __restrict__ table, const int* __restrict__ rpi,
                   float* __restrict__ bias) {
  int idx = blockIdx.x * 256 + threadIdx.x;   // < 4*64*64
  int h = idx >> 12;
  int qk = idx & 4095;
  float bv = table[rpi[qk] * 4 + h];
  bias[idx] = 16.f / (1.f + __expf(-bv));     // 16 * sigmoid
}

// --------------------------------------------------------- fused window attention
__global__ __launch_bounds__(256)
void k_win_attn(const float* __restrict__ hsw, const float* __restrict__ q_in,
                const float* __restrict__ k_in, const _Float16* __restrict__ vw16,
                const float* __restrict__ value_b, const _Float16* __restrict__ ow16,
                const float* __restrict__ out_b, const float* __restrict__ logit_scale,
                const float* __restrict__ bias_g, float* __restrict__ x_out) {
  __shared__ __align__(32) _Float16 sA[64 * 128];        // V^T, then ctx
  __shared__ __align__(32) _Float16 sB[2 * 4 * 64 * 32]; // Q|K, then P
  _Float16* sQ  = sB;
  _Float16* sK  = sB + 4 * 64 * 32;
  _Float16* sP  = sB;        // aliases Q|K (after barrier)
  _Float16* sVt = sA;        // [128][64] channel-major
  _Float16* sCtx = sA;       // [64][128] (after barrier)

  const int tid  = threadIdx.x;
  const int lane = tid & 31;
  const int wv   = tid >> 5;
  const int w    = blockIdx.x;
  const int l15  = lane & 15;
  const int lh   = lane >> 4;
  const size_t hs_base = (size_t)w * WW_ * C_;

  // ---- Phase A: V = hs @ value_w^T + value_b, stored transposed sVt[n][m]
  {
    const int n0 = wv * 16;
#pragma unroll
    for (int mt = 0; mt < 4; ++mt) {
      v8f acc = {};
#pragma unroll
      for (int kk = 0; kk < 4; ++kk) {
        int kh = kk * 32 + lh * 16;
        v16h a = frag_f32(hsw + hs_base + (size_t)(mt * 16 + l15) * C_ + kh);
        v16h b = frag_h(vw16 + (size_t)(n0 + l15) * C_ + kh);
        acc = wmma_f16(a, b, acc);
      }
#pragma unroll
      for (int i = 0; i < 8; ++i) {
        int m = mt * 16 + i + lh * 8;
        int n = n0 + l15;
        sVt[n * 64 + m] = (_Float16)(acc[i] + value_b[n]);
      }
    }
  }

  // ---- Phase B: row-normalize q (fold per-head scale) and k -> LDS f16
  {
    const int h = tid >> 6;
    const int t = tid & 63;
    const float sc = __expf(fminf(logit_scale[h], 4.6051702f)); // log(100)
    const size_t base = ((size_t)w * NH_ + h) * WW_ * HD_ + (size_t)t * HD_;
    float qv[32], kv[32];
    float qs = 0.f, ks = 0.f;
#pragma unroll
    for (int d = 0; d < 32; ++d) { qv[d] = q_in[base + d]; qs += qv[d] * qv[d]; }
#pragma unroll
    for (int d = 0; d < 32; ++d) { kv[d] = k_in[base + d]; ks += kv[d] * kv[d]; }
    float qr = sc / fmaxf(sqrtf(qs), 1e-12f);
    float kr = 1.f / fmaxf(sqrtf(ks), 1e-12f);
    _Float16* qd = sQ + ((size_t)h * 64 + t) * 32;
    _Float16* kd = sK + ((size_t)h * 64 + t) * 32;
#pragma unroll
    for (int d = 0; d < 32; ++d) {
      qd[d] = (_Float16)(qv[d] * qr);
      kd[d] = (_Float16)(kv[d] * kr);
    }
  }
  __syncthreads();

  // ---- Phase C: S = Qn Kn^T + bias, softmax (row-wise, in registers)
  v8f p_acc[2][4];
  int p_h[2], p_mt[2];
#pragma unroll
  for (int pp = 0; pp < 2; ++pp) {
    int pr = wv * 2 + pp;
    int h = pr >> 2, mt = pr & 3;
    p_h[pp] = h; p_mt[pp] = mt;
    v16h a = frag_h(sQ + ((size_t)h * 64 + mt * 16 + l15) * 32 + lh * 16);
#pragma unroll
    for (int nt = 0; nt < 4; ++nt) {
      v16h b = frag_h(sK + ((size_t)h * 64 + nt * 16 + l15) * 32 + lh * 16);
      v8f acc = {};
      acc = wmma_f16(a, b, acc);
#pragma unroll
      for (int i = 0; i < 8; ++i) {
        int r = mt * 16 + i + lh * 8;
        int c = nt * 16 + l15;
        acc[i] += bias_g[((size_t)h * 64 + r) * 64 + c];
      }
      p_acc[pp][nt] = acc;
    }
#pragma unroll
    for (int i = 0; i < 8; ++i) {
      float mx = -3.0e38f;
#pragma unroll
      for (int nt = 0; nt < 4; ++nt) mx = fmaxf(mx, p_acc[pp][nt][i]);
#pragma unroll
      for (int msk = 8; msk >= 1; msk >>= 1) mx = fmaxf(mx, __shfl_xor(mx, msk, 32));
      float s = 0.f;
#pragma unroll
      for (int nt = 0; nt < 4; ++nt) {
        float e = __expf(p_acc[pp][nt][i] - mx);
        p_acc[pp][nt][i] = e;
        s += e;
      }
#pragma unroll
      for (int msk = 8; msk >= 1; msk >>= 1) s += __shfl_xor(s, msk, 32);
      float inv = 1.f / s;
#pragma unroll
      for (int nt = 0; nt < 4; ++nt) p_acc[pp][nt][i] *= inv;
    }
  }
  __syncthreads();                 // done reading sQ/sK everywhere
#pragma unroll
  for (int pp = 0; pp < 2; ++pp) {
    int h = p_h[pp], mt = p_mt[pp];
#pragma unroll
    for (int nt = 0; nt < 4; ++nt)
#pragma unroll
      for (int i = 0; i < 8; ++i) {
        int r = mt * 16 + i + lh * 8;
        int c = nt * 16 + l15;
        sP[((size_t)h * 64 + r) * 64 + c] = (_Float16)p_acc[pp][nt][i];
      }
  }
  __syncthreads();

  // ---- Phase D: ctx = P @ V  (B operand is sVt, channel-major = V^T)
  v8f c_acc[4];
  int c_h[4], c_mt[4], c_dt[4];
#pragma unroll
  for (int q = 0; q < 4; ++q) {
    int idx = wv * 4 + q;
    int h = idx >> 3, rem = idx & 7;
    int mt = rem >> 1, dt = rem & 1;
    c_h[q] = h; c_mt[q] = mt; c_dt[q] = dt;
    v8f acc = {};
#pragma unroll
    for (int kk = 0; kk < 2; ++kk) {
      int kh = kk * 32 + lh * 16;
      v16h a = frag_h(sP + ((size_t)h * 64 + mt * 16 + l15) * 64 + kh);
      v16h b = frag_h(sVt + (size_t)(h * 32 + dt * 16 + l15) * 64 + kh);
      acc = wmma_f16(a, b, acc);
    }
    c_acc[q] = acc;
  }
  __syncthreads();                 // done reading sP/sVt
#pragma unroll
  for (int q = 0; q < 4; ++q) {
    int h = c_h[q], mt = c_mt[q], dt = c_dt[q];
#pragma unroll
    for (int i = 0; i < 8; ++i) {
      int m = mt * 16 + i + lh * 8;
      int n = h * 32 + dt * 16 + l15;
      sCtx[(size_t)m * 128 + n] = (_Float16)c_acc[q][i];
    }
  }
  __syncthreads();

  // ---- Phase E: out = ctx @ out_w^T + out_b, window-reverse scatter to x
  {
    const int n0 = wv * 16;
    const int b_img = w >> 6;
    const int wr = (w >> 3) & 7;
    const int wc = w & 7;
#pragma unroll
    for (int mt = 0; mt < 4; ++mt) {
      v8f acc = {};
#pragma unroll
      for (int kk = 0; kk < 4; ++kk) {
        int kh = kk * 32 + lh * 16;
        v16h a = frag_h(sCtx + (size_t)(mt * 16 + l15) * 128 + kh);
        v16h b = frag_h(ow16 + (size_t)(n0 + l15) * C_ + kh);
        acc = wmma_f16(a, b, acc);
      }
#pragma unroll
      for (int i = 0; i < 8; ++i) {
        int m = mt * 16 + i + lh * 8;
        int n = n0 + l15;
        int tr = m >> 3, tc = m & 7;
        int pos = (wr * 8 + tr) * 64 + (wc * 8 + tc);
        x_out[((size_t)b_img * HW_ + pos) * C_ + n] = acc[i] + out_b[n];
      }
    }
  }
}

// ------------------------------------------- hs = emb + LN_before(x)  (wave/row)
__global__ __launch_bounds__(256)
void k_res_ln(const float* __restrict__ x, const float* __restrict__ emb,
              const float* __restrict__ lw, const float* __restrict__ lb,
              float* __restrict__ out) {
  const size_t row = (size_t)blockIdx.x * 8 + (threadIdx.x >> 5);
  const int lane = threadIdx.x & 31;
  float4 v = ((const float4*)(x + row * 128))[lane];
  float s  = v.x + v.y + v.z + v.w;
  float ss = v.x * v.x + v.y * v.y + v.z * v.z + v.w * v.w;
#pragma unroll
  for (int m = 16; m >= 1; m >>= 1) { s += __shfl_xor(s, m, 32); ss += __shfl_xor(ss, m, 32); }
  float mu = s * (1.f / 128.f);
  float var = ss * (1.f / 128.f) - mu * mu;
  float rs = rsqrtf(var + LN_EPS);
  int c0 = lane * 4;
  float4 e = ((const float4*)(emb + row * 128))[lane];
  float4 o;
  o.x = e.x + (v.x - mu) * rs * lw[c0 + 0] + lb[c0 + 0];
  o.y = e.y + (v.y - mu) * rs * lw[c0 + 1] + lb[c0 + 1];
  o.z = e.z + (v.z - mu) * rs * lw[c0 + 2] + lb[c0 + 2];
  o.w = e.w + (v.w - mu) * rs * lw[c0 + 3] + lb[c0 + 3];
  ((float4*)(out + row * 128))[lane] = o;
}

// -------------------------------------- FFN1: gelu(hs @ inter_w^T + b) -> fp16
// A tile staged global -> LDS through the CDNA5 async-LDS path (ASYNCcnt).
__global__ __launch_bounds__(256)
void k_ffn1(const float* __restrict__ hs, const _Float16* __restrict__ iw16,
            const float* __restrict__ ib, _Float16* __restrict__ o) {
  __shared__ __align__(16) float sAf[64 * 128];   // fp32 tile, 32KB
  const int tid = threadIdx.x, lane = tid & 31, wv = tid >> 5;
  const int l15 = lane & 15, lh = lane >> 4;
  const int n0 = blockIdx.x * 64;
  const size_t m0 = (size_t)blockIdx.y * 64;

  // async copy 32KB: 8 iters x 256 lanes x 16B  (LDS addr = low 32 bits of
  // the generic shared pointer = offset within the wave's LDS allocation)
  {
    const float* gsrc = hs + m0 * 128 + tid * 4;
    unsigned laddr = (unsigned)(size_t)(void*)(sAf + tid * 4);
#pragma unroll
    for (int it = 0; it < 8; ++it) {
      asm volatile("global_load_async_to_lds_b128 %0, %1, off"
                   :: "v"(laddr + it * 4096u), "v"(gsrc + it * 1024)
                   : "memory");
    }
    asm volatile("s_wait_asynccnt 0x0" ::: "memory");
  }
  __syncthreads();

#pragma unroll
  for (int t = 0; t < 2; ++t) {
    int idx = wv * 2 + t;
    int mt = idx >> 2, nt = idx & 3;
    v8f acc = {};
#pragma unroll
    for (int kk = 0; kk < 4; ++kk) {
      int kh = kk * 32 + lh * 16;
      v16h a = frag_f32(sAf + (size_t)(mt * 16 + l15) * 128 + kh);
      v16h b = frag_h(iw16 + (size_t)(n0 + nt * 16 + l15) * 128 + kh);
      acc = wmma_f16(a, b, acc);
    }
#pragma unroll
    for (int i = 0; i < 8; ++i) {
      int m = mt * 16 + i + lh * 8;
      int n = n0 + nt * 16 + l15;
      float vv = acc[i] + ib[n];
      float g = 0.5f * vv * (1.f + erff(vv * 0.70710678f));  // exact gelu
      o[(m0 + m) * FF_ + n] = (_Float16)g;
    }
  }
}

// ---------------- FFN2: layer_out = hs + LN_after(ffn1 @ ffn_out_w^T + b) fused
__global__ __launch_bounds__(256)
void k_ffn2(const _Float16* __restrict__ a1, const _Float16* __restrict__ ow16,
            const float* __restrict__ ob, const float* __restrict__ hs,
            const float* __restrict__ lw, const float* __restrict__ lb,
            float* __restrict__ out) {
  __shared__ __align__(16) float sO[64 * 128];
  const int tid = threadIdx.x, lane = tid & 31, wv = tid >> 5;
  const int l15 = lane & 15, lh = lane >> 4;
  const size_t m0 = (size_t)blockIdx.x * 64;
  __builtin_prefetch(hs + (m0 + (size_t)(tid >> 2)) * 128, 0, 3);
#pragma unroll
  for (int q = 0; q < 4; ++q) {
    int idx = wv * 4 + q;
    int mt = idx >> 3, nt = idx & 7;
    v8f acc = {};
#pragma unroll
    for (int kk = 0; kk < 16; ++kk) {
      int kh = kk * 32 + lh * 16;
      v16h a = frag_h(a1 + (m0 + mt * 16 + l15) * FF_ + kh);
      v16h b = frag_h(ow16 + (size_t)(nt * 16 + l15) * FF_ + kh);
      acc = wmma_f16(a, b, acc);
    }
#pragma unroll
    for (int i = 0; i < 8; ++i) {
      int m = mt * 16 + i + lh * 8;
      int n = nt * 16 + l15;
      sO[m * 128 + n] = acc[i] + ob[n];
    }
  }
  __syncthreads();
#pragma unroll
  for (int it = 0; it < 8; ++it) {
    int r = wv + it * 8;
    float4 v = ((const float4*)(sO + r * 128))[lane];
    float s  = v.x + v.y + v.z + v.w;
    float ss = v.x * v.x + v.y * v.y + v.z * v.z + v.w * v.w;
#pragma unroll
    for (int m = 16; m >= 1; m >>= 1) { s += __shfl_xor(s, m, 32); ss += __shfl_xor(ss, m, 32); }
    float mu = s * (1.f / 128.f);
    float var = ss * (1.f / 128.f) - mu * mu;
    float rs = rsqrtf(var + LN_EPS);
    size_t rg = m0 + r;
    int c0 = lane * 4;
    float4 h4 = ((const float4*)(hs + rg * 128))[lane];
    float4 o;
    o.x = h4.x + (v.x - mu) * rs * lw[c0 + 0] + lb[c0 + 0];
    o.y = h4.y + (v.y - mu) * rs * lw[c0 + 1] + lb[c0 + 1];
    o.z = h4.z + (v.z - mu) * rs * lw[c0 + 2] + lb[c0 + 2];
    o.w = h4.w + (v.w - mu) * rs * lw[c0 + 3] + lb[c0 + 3];
    ((float4*)(out + rg * 128))[lane] = o;
  }
}

// -------- patch merge: LN(gather4 @ ds_red_w^T), gather fused into A fragments
__global__ __launch_bounds__(256)
void k_merge(const float* __restrict__ lo, const _Float16* __restrict__ rw16,
             const float* __restrict__ nw, const float* __restrict__ nb,
             float* __restrict__ out) {
  __shared__ __align__(16) float sO[32 * 256];
  const int tid = threadIdx.x, lane = tid & 31, wv = tid >> 5;
  const int l15 = lane & 15, lh = lane >> 4;
  const int m0 = blockIdx.x * 32;
#pragma unroll
  for (int q = 0; q < 4; ++q) {
    int idx = wv * 4 + q;
    int mt = idx >> 4, nt = idx & 15;
    int m = m0 + mt * 16 + l15;
    int b = m >> 10;
    int ij = m & 1023;
    int i = ij >> 5, j = ij & 31;
    v8f acc = {};
#pragma unroll
    for (int kk = 0; kk < 16; ++kk) {
      int kh = kk * 32 + lh * 16;
      int qd = kh >> 7;            // quadrant: 0:(0,0) 1:(1,0) 2:(0,1) 3:(1,1)
      int kin = kh & 127;
      size_t src = ((size_t)b * HW_ +
                    (size_t)(2 * i + (qd & 1)) * 64 + (2 * j + (qd >> 1))) * 128 + kin;
      v16h a  = frag_f32(lo + src);
      v16h bb = frag_h(rw16 + (size_t)(nt * 16 + l15) * FF_ + kh);
      acc = wmma_f16(a, bb, acc);
    }
#pragma unroll
    for (int ii = 0; ii < 8; ++ii) {
      int mm = mt * 16 + ii + lh * 8;
      int n = nt * 16 + l15;
      sO[mm * 256 + n] = acc[ii];
    }
  }
  __syncthreads();
#pragma unroll
  for (int it = 0; it < 4; ++it) {
    int r = wv * 4 + it;
    const float4* pr = (const float4*)(sO + r * 256);
    float4 a0 = pr[lane * 2], a1 = pr[lane * 2 + 1];
    float s  = a0.x + a0.y + a0.z + a0.w + a1.x + a1.y + a1.z + a1.w;
    float ss = a0.x*a0.x + a0.y*a0.y + a0.z*a0.z + a0.w*a0.w +
               a1.x*a1.x + a1.y*a1.y + a1.z*a1.z + a1.w*a1.w;
#pragma unroll
    for (int m = 16; m >= 1; m >>= 1) { s += __shfl_xor(s, m, 32); ss += __shfl_xor(ss, m, 32); }
    float mu = s * (1.f / 256.f);
    float var = ss * (1.f / 256.f) - mu * mu;
    float rs = rsqrtf(var + LN_EPS);
    float vals[8] = {a0.x, a0.y, a0.z, a0.w, a1.x, a1.y, a1.z, a1.w};
    int c0 = lane * 8;
    float* orow = out + (size_t)(m0 + r) * 256;
#pragma unroll
    for (int jj = 0; jj < 8; ++jj)
      orow[c0 + jj] = (vals[jj] - mu) * rs * nw[c0 + jj] + nb[c0 + jj];
  }
}

// ---------------------------------------------------------------------- launch
extern "C" void kernel_launch(void* const* d_in, const int* in_sizes, int n_in,
                              void* d_out, int out_size, void* d_ws, size_t ws_size,
                              hipStream_t stream) {
  (void)in_sizes; (void)n_in; (void)out_size; (void)ws_size;
  const float* out_dense_b = (const float*)d_in[0];
  const float* out_dense_w = (const float*)d_in[1];
  const float* rct         = (const float*)d_in[2];
  const float* cpb0_b      = (const float*)d_in[3];
  const float* cpb0_w      = (const float*)d_in[4];
  const float* cpb2_w      = (const float*)d_in[5];
  const float* value_b     = (const float*)d_in[6];
  const float* value_w     = (const float*)d_in[7];
  const float* logit_scale = (const float*)d_in[8];
  const float* inter_b     = (const float*)d_in[9];
  const float* inter_w     = (const float*)d_in[10];
  const float* ln_after_b  = (const float*)d_in[11];
  const float* ln_after_w  = (const float*)d_in[12];
  const float* ln_before_b = (const float*)d_in[13];
  const float* ln_before_w = (const float*)d_in[14];
  const float* ffn_out_b   = (const float*)d_in[15];
  const float* ffn_out_w   = (const float*)d_in[16];
  const float* ds_norm_b   = (const float*)d_in[17];
  const float* ds_norm_w   = (const float*)d_in[18];
  const float* ds_red_w    = (const float*)d_in[19];
  const float* emb3        = (const float*)d_in[20];
  const float* hsw         = (const float*)d_in[21];
  const float* k_in        = (const float*)d_in[22];
  const float* q_in        = (const float*)d_in[23];
  const int*   rpi         = (const int*)d_in[24];

  char* ws = (char*)d_ws;
  size_t off = 0;
  auto carve = [&](size_t bytes) -> void* {
    void* p = ws + off;
    off += (bytes + 255) & ~(size_t)255;
    return p;
  };
  float*    bias_table = (float*)carve(225 * 4 * sizeof(float));
  float*    bias_g     = (float*)carve((size_t)NH_ * WW_ * WW_ * sizeof(float));
  float*    x_buf      = (float*)carve((size_t)NTOK * C_ * sizeof(float)); // attn-out, later layer_out
  float*    hs_buf     = (float*)carve((size_t)NTOK * C_ * sizeof(float));
  _Float16* ffn1       = (_Float16*)carve((size_t)NTOK * FF_ * sizeof(_Float16));
  _Float16* vw16       = (_Float16*)carve((size_t)C_ * C_ * 2);
  _Float16* ow16       = (_Float16*)carve((size_t)C_ * C_ * 2);
  _Float16* iw16       = (_Float16*)carve((size_t)FF_ * C_ * 2);
  _Float16* fw16       = (_Float16*)carve((size_t)C_ * FF_ * 2);
  _Float16* rw16       = (_Float16*)carve((size_t)2 * C_ * FF_ * 2);

  // one-time fp32 -> fp16 weight casts (tiny; keeps GEMM loops convert-free)
  k_cvt_f16<<<16,  256, 0, stream>>>((const float4*)value_w,   (v4h*)vw16);
  k_cvt_f16<<<16,  256, 0, stream>>>((const float4*)out_dense_w,(v4h*)ow16);
  k_cvt_f16<<<64,  256, 0, stream>>>((const float4*)inter_w,   (v4h*)iw16);
  k_cvt_f16<<<64,  256, 0, stream>>>((const float4*)ffn_out_w, (v4h*)fw16);
  k_cvt_f16<<<128, 256, 0, stream>>>((const float4*)ds_red_w,  (v4h*)rw16);

  k_cpb_table<<<225, 256, 0, stream>>>(rct, cpb0_w, cpb0_b, cpb2_w, bias_table);
  k_bias_expand<<<64, 256, 0, stream>>>(bias_table, rpi, bias_g);
  k_win_attn<<<BW_, 256, 0, stream>>>(hsw, q_in, k_in, vw16, value_b,
                                      ow16, out_dense_b, logit_scale,
                                      bias_g, x_buf);
  k_res_ln<<<NTOK / 8, 256, 0, stream>>>(x_buf, emb3, ln_before_w, ln_before_b, hs_buf);
  k_ffn1<<<dim3(8, NTOK / 64), 256, 0, stream>>>(hs_buf, iw16, inter_b, ffn1);
  k_ffn2<<<NTOK / 64, 256, 0, stream>>>(ffn1, fw16, ffn_out_b, hs_buf,
                                        ln_after_w, ln_after_b, x_buf);
  k_merge<<<NTOK / 4 / 32, 256, 0, stream>>>(x_buf, rw16, ds_norm_w, ds_norm_b,
                                             (float*)d_out);
}
